// MultiheadSelfAttention_84181359001857
// MI455X (gfx1250) — compile-verified, matrix operand order checked by reading, not run on hardware
//
#include <hip/hip_runtime.h>
#include <hip/hip_bf16.h>
#include <stdint.h>

// ---------------- problem constants ----------------
constexpr int Bc = 2;
constexpr int Sc = 2048;
constexpr int Dc = 1024;
constexpr int Hc = 16;
constexpr int DKc = 64;          // Dc / Hc
constexpr int Mrows = Bc * Sc;   // 4096

typedef _Float16 half4_t __attribute__((ext_vector_type(4)));
typedef _Float16 half8_t __attribute__((ext_vector_type(8)));
typedef _Float16 v16h    __attribute__((ext_vector_type(16)));
typedef float    v8f     __attribute__((ext_vector_type(8)));

__device__ __forceinline__ v16h mk16(half8_t lo, half8_t hi) {
    v16h r;
#pragma unroll
    for (int j = 0; j < 8; ++j) { r[j] = lo[j]; r[8 + j] = hi[j]; }
    return r;
}

// Async global->LDS 16-byte copy (CDNA5 ASYNCcnt path).
__device__ __forceinline__ void async_copy_b128(uint32_t lds_addr, uint64_t gaddr) {
    asm volatile("global_load_async_to_lds_b128 %0, %1, off"
                 :: "v"(lds_addr), "v"(gaddr) : "memory");
}
__device__ __forceinline__ void wait_async0() {
    asm volatile("s_wait_asynccnt 0x0" ::: "memory");
}

// =====================================================================
// f32 -> f16 bulk convert (one-time pass so GEMM tiles are byte copies)
// =====================================================================
__global__ __launch_bounds__(256) void cvt_f32_f16(const float* __restrict__ in,
                                                   _Float16* __restrict__ out,
                                                   int n4) {
    int i = blockIdx.x * blockDim.x + threadIdx.x;
    if (i < n4) {
        float4 v = ((const float4*)in)[i];
        half4_t h = {(_Float16)v.x, (_Float16)v.y, (_Float16)v.z, (_Float16)v.w};
        ((half4_t*)out)[i] = h;
    }
}

// =====================================================================
// GEMM: Out[m][n] = sum_k A[m][k] * W[n][k]   (A,W f16; WMMA f32 acc).
// Block tile 128x128, 8 waves, wave tile 32x64.
// Double-buffered LDS staging via async global->LDS copies.
// =====================================================================
template <bool OUT_F16>
__global__ __launch_bounds__(256) void gemm_h(const _Float16* __restrict__ A,
                                              const _Float16* __restrict__ W,
                                              void* __restrict__ Out) {
    __shared__ _Float16 At[2][128][32];
    __shared__ _Float16 Bt[2][128][32];

    const int t    = threadIdx.x;
    const int w    = t >> 5;
    const int lane = t & 31;
    const int hf   = lane >> 4;
    const int ln   = lane & 15;
    const int wm   = w & 3;        // 4 wave-rows of 32
    const int wn   = w >> 2;       // 2 wave-cols of 64
    const int mb   = blockIdx.x * 128;
    const int nb   = blockIdx.y * 128;

    auto issue_tiles = [&](int buf, int k0) {
#pragma unroll
        for (int c = 0; c < 2; ++c) {
            int chunk = t + 256 * c;          // 0..511 : 128 rows x 4 chunks
            int row   = chunk >> 2;
            int c8    = (chunk & 3) << 3;     // f16 col: 0,8,16,24
            async_copy_b128((uint32_t)(uintptr_t)&At[buf][row][c8],
                            (uint64_t)(uintptr_t)&A[(size_t)(mb + row) * Dc + k0 + c8]);
            async_copy_b128((uint32_t)(uintptr_t)&Bt[buf][row][c8],
                            (uint64_t)(uintptr_t)&W[(size_t)(nb + row) * Dc + k0 + c8]);
        }
    };

    v8f acc[2][4];
#pragma unroll
    for (int mi = 0; mi < 2; ++mi)
#pragma unroll
        for (int ni = 0; ni < 4; ++ni) acc[mi][ni] = (v8f)(0.0f);

    issue_tiles(0, 0);
    wait_async0();
    __syncthreads();

    for (int ks = 0; ks < Dc / 32; ++ks) {
        const int cur = ks & 1;
        if (ks + 1 < Dc / 32) issue_tiles(cur ^ 1, (ks + 1) * 32);   // overlap copy

        v16h af[2];
#pragma unroll
        for (int mi = 0; mi < 2; ++mi) {
            int r = wm * 32 + mi * 16 + ln;
            half8_t lo = *(const half8_t*)&At[cur][r][hf * 8];
            half8_t hi = *(const half8_t*)&At[cur][r][16 + hf * 8];
            af[mi] = mk16(lo, hi);
        }
        v16h bf[4];
#pragma unroll
        for (int ni = 0; ni < 4; ++ni) {
            int r = wn * 64 + ni * 16 + ln;
            half8_t lo = *(const half8_t*)&Bt[cur][r][hf * 8];
            half8_t hi = *(const half8_t*)&Bt[cur][r][16 + hf * 8];
            bf[ni] = mk16(lo, hi);
        }
#pragma unroll
        for (int mi = 0; mi < 2; ++mi)
#pragma unroll
            for (int ni = 0; ni < 4; ++ni)
                acc[mi][ni] = __builtin_amdgcn_wmma_f32_16x16x32_f16(
                    false, af[mi], false, bf[ni], (short)0, acc[mi][ni], false, false);

        wait_async0();
        __syncthreads();
    }

#pragma unroll
    for (int mi = 0; mi < 2; ++mi)
#pragma unroll
        for (int ni = 0; ni < 4; ++ni)
#pragma unroll
            for (int r = 0; r < 8; ++r) {
                size_t row = (size_t)(mb + wm * 32 + mi * 16 + hf * 8 + r);
                size_t col = (size_t)(nb + wn * 64 + ni * 16 + ln);
                if (OUT_F16)
                    ((_Float16*)Out)[row * Dc + col] = (_Float16)acc[mi][ni][r];
                else
                    ((float*)Out)[row * Dc + col] = acc[mi][ni][r];
            }
}

// =====================================================================
// RoPE (in place on f16 Q or K). One thread per even/odd pair.
// =====================================================================
__global__ __launch_bounds__(256) void rope_f16(_Float16* __restrict__ X,
                                                const int* __restrict__ pos,
                                                int npairs) {
    int p = blockIdx.x * blockDim.x + threadIdx.x;
    if (p >= npairs) return;
    int i   = p & 31;            // pair index within head (DK/2 = 32)
    int hh  = (p >> 5) & 15;     // head
    int row = p >> 9;            // b*S + s
    int s   = row & (Sc - 1);
    float ang = (float)pos[s] * __powf(10000.0f, -(float)i / 32.0f);
    float sn, cs;
    __sincosf(ang, &sn, &cs);
    size_t base = (size_t)row * Dc + hh * DKc + 2 * i;
    float x1 = (float)X[base];
    float x2 = (float)X[base + 1];
    X[base]     = (_Float16)(cs * x1 - sn * x2);
    X[base + 1] = (_Float16)(sn * x1 + cs * x2);
}

// =====================================================================
// Flash attention: grid (S/64, H, B), 128 threads (4 waves x 16 q rows).
// Double-buffered K tile via async global->LDS; V transposed on the
// synchronous path (overlaps with the async copy). Output written as f16.
// =====================================================================
__global__ __launch_bounds__(128) void flash_attn(const _Float16* __restrict__ Q,
                                                  const _Float16* __restrict__ K,
                                                  const _Float16* __restrict__ V,
                                                  _Float16* __restrict__ O) {
    __shared__ _Float16 Kt[2][64][64];    // [key][dk] row-major
    __shared__ _Float16 Vt[2][64][64];    // [dk][key] transposed
    __shared__ _Float16 Pt[4][16][64];    // per-wave P tile

    const int t    = threadIdx.x;
    const int w    = t >> 5;
    const int lane = t & 31;
    const int hf   = lane >> 4;
    const int ln   = lane & 15;
    const int qb   = blockIdx.x;
    const int h    = blockIdx.y;
    const int b    = blockIdx.z;

    auto stage_kv = [&](int buf, int kb) {
#pragma unroll
        for (int c = 0; c < 4; ++c) {
            int chunk = t + 128 * c;          // 0..511
            int key   = chunk >> 3;
            int dk0   = (chunk & 7) << 3;
            async_copy_b128((uint32_t)(uintptr_t)&Kt[buf][key][dk0],
                            (uint64_t)(uintptr_t)&K[(size_t)(b * Sc + kb * 64 + key) * Dc + h * DKc + dk0]);
        }
#pragma unroll
        for (int c = 0; c < 4; ++c) {
            int chunk = t + 128 * c;
            int key   = chunk >> 3;
            int dk0   = (chunk & 7) << 3;
            half8_t vv = *(const half8_t*)&V[(size_t)(b * Sc + kb * 64 + key) * Dc + h * DKc + dk0];
#pragma unroll
            for (int j = 0; j < 8; ++j) Vt[buf][dk0 + j][key] = vv[j];
        }
    };

    // ---- resident Q fragments for this wave's 16 rows ----
    const size_t qrowbase = (size_t)(b * Sc + qb * 64 + w * 16 + ln) * Dc + h * DKc;
    v16h qf[2];
#pragma unroll
    for (int kf = 0; kf < 2; ++kf) {
        half8_t lo = *(const half8_t*)&Q[qrowbase + kf * 32 + hf * 8];
        half8_t hi = *(const half8_t*)&Q[qrowbase + kf * 32 + 16 + hf * 8];
        qf[kf] = mk16(lo, hi);
    }

    float mrow[8], lrow[8];
    v8f   Oacc[4];
#pragma unroll
    for (int r = 0; r < 8; ++r) { mrow[r] = -INFINITY; lrow[r] = 0.0f; }
#pragma unroll
    for (int nt = 0; nt < 4; ++nt) Oacc[nt] = (v8f)(0.0f);

    stage_kv(0, 0);
    wait_async0();
    __syncthreads();

    for (int kb = 0; kb <= qb; ++kb) {
        const int cur = kb & 1;
        if (kb < qb) stage_kv(cur ^ 1, kb + 1);     // overlap next tile copy

        // ---- S = Q K^T / sqrt(dk), causal mask on diagonal block ----
        float P[4][8];
        float bmax[8];
#pragma unroll
        for (int r = 0; r < 8; ++r) bmax[r] = -INFINITY;
#pragma unroll
        for (int nt = 0; nt < 4; ++nt) {
            v8f s = (v8f)(0.0f);
#pragma unroll
            for (int kf = 0; kf < 2; ++kf) {
                half8_t lo = *(const half8_t*)&Kt[cur][nt * 16 + ln][kf * 32 + hf * 8];
                half8_t hi = *(const half8_t*)&Kt[cur][nt * 16 + ln][kf * 32 + 16 + hf * 8];
                v16h bfrag = mk16(lo, hi);
                s = __builtin_amdgcn_wmma_f32_16x16x32_f16(false, qf[kf], false, bfrag,
                                                           (short)0, s, false, false);
            }
#pragma unroll
            for (int r = 0; r < 8; ++r) {
                float v = s[r] * 0.125f;          // 1/sqrt(64)
                if (kb == qb) {
                    int qrl = w * 16 + hf * 8 + r;
                    int krl = nt * 16 + ln;
                    if (krl > qrl) v = -INFINITY;
                }
                P[nt][r] = v;
                bmax[r]  = fmaxf(bmax[r], v);
            }
        }
        // ---- row max across the 16 lanes holding one row ----
#pragma unroll
        for (int r = 0; r < 8; ++r) {
#pragma unroll
            for (int off = 1; off < 16; off <<= 1)
                bmax[r] = fmaxf(bmax[r], __shfl_xor(bmax[r], off, 32));
        }
        float sc[8];
#pragma unroll
        for (int r = 0; r < 8; ++r) {
            float mn = fmaxf(mrow[r], bmax[r]);
            sc[r]    = __expf(mrow[r] - mn);
            mrow[r]  = mn;
        }
        float rs[8];
#pragma unroll
        for (int r = 0; r < 8; ++r) rs[r] = 0.0f;
#pragma unroll
        for (int nt = 0; nt < 4; ++nt)
#pragma unroll
            for (int r = 0; r < 8; ++r) {
                P[nt][r] = __expf(P[nt][r] - mrow[r]);
                rs[r] += P[nt][r];
            }
#pragma unroll
        for (int r = 0; r < 8; ++r) {
#pragma unroll
            for (int off = 1; off < 16; off <<= 1)
                rs[r] += __shfl_xor(rs[r], off, 32);
            lrow[r] = lrow[r] * sc[r] + rs[r];
        }
#pragma unroll
        for (int nt = 0; nt < 4; ++nt)
#pragma unroll
            for (int r = 0; r < 8; ++r) Oacc[nt][r] *= sc[r];

        // ---- bounce P through LDS into A-fragment layout ----
#pragma unroll
        for (int nt = 0; nt < 4; ++nt)
#pragma unroll
            for (int r = 0; r < 8; ++r)
                Pt[w][hf * 8 + r][nt * 16 + ln] = (_Float16)P[nt][r];
        asm volatile("s_wait_dscnt 0x0" ::: "memory");   // same-wave LDS RAW

        v16h pf[2];
#pragma unroll
        for (int kf = 0; kf < 2; ++kf) {
            half8_t lo = *(const half8_t*)&Pt[w][ln][kf * 32 + hf * 8];
            half8_t hi = *(const half8_t*)&Pt[w][ln][kf * 32 + 16 + hf * 8];
            pf[kf] = mk16(lo, hi);
        }
        // ---- O += P @ V ----
#pragma unroll
        for (int nt = 0; nt < 4; ++nt) {
#pragma unroll
            for (int kf = 0; kf < 2; ++kf) {
                half8_t lo = *(const half8_t*)&Vt[cur][nt * 16 + ln][kf * 32 + hf * 8];
                half8_t hi = *(const half8_t*)&Vt[cur][nt * 16 + ln][kf * 32 + 16 + hf * 8];
                v16h vfrag = mk16(lo, hi);
                Oacc[nt] = __builtin_amdgcn_wmma_f32_16x16x32_f16(
                    false, pf[kf], false, vfrag, (short)0, Oacc[nt], false, false);
            }
        }

        wait_async0();
        __syncthreads();
    }

    // ---- normalize, write attention output as f16 [B,S,D] ----
#pragma unroll
    for (int r = 0; r < 8; ++r) {
        float inv = 1.0f / lrow[r];
#pragma unroll
        for (int nt = 0; nt < 4; ++nt) {
            size_t row = (size_t)(b * Sc + qb * 64 + w * 16 + hf * 8 + r);
            O[row * Dc + h * DKc + nt * 16 + ln] = (_Float16)(Oacc[nt][r] * inv);
        }
    }
}

// =====================================================================
// launch
// =====================================================================
extern "C" void kernel_launch(void* const* d_in, const int* in_sizes, int n_in,
                              void* d_out, int out_size, void* d_ws, size_t ws_size,
                              hipStream_t stream) {
    const float* x   = (const float*)d_in[0];
    const int*   pos = (const int*)d_in[1];
    const float* WQ  = (const float*)d_in[2];
    const float* WK  = (const float*)d_in[3];
    const float* WV  = (const float*)d_in[4];
    const float* WO  = (const float*)d_in[5];

    const size_t szX = (size_t)Bc * Sc * Dc;   // 4M elements
    const size_t szW = (size_t)Dc * Dc;        // 1M elements

    _Float16* Xh  = (_Float16*)d_ws;
    _Float16* WQh = Xh + szX;
    _Float16* WKh = WQh + szW;
    _Float16* WVh = WKh + szW;
    _Float16* WOh = WVh + szW;
    _Float16* Qh  = WOh + szW;
    _Float16* Kh  = Qh + szX;
    _Float16* Vh  = Kh + szX;
    _Float16* Oh  = Vh + szX;

    // ---- one-time f32 -> f16 conversions ----
    int nx4 = (int)(szX / 4), nw4 = (int)(szW / 4);
    cvt_f32_f16<<<(nx4 + 255) / 256, 256, 0, stream>>>(x, Xh, nx4);
    cvt_f32_f16<<<(nw4 + 255) / 256, 256, 0, stream>>>(WQ, WQh, nw4);
    cvt_f32_f16<<<(nw4 + 255) / 256, 256, 0, stream>>>(WK, WKh, nw4);
    cvt_f32_f16<<<(nw4 + 255) / 256, 256, 0, stream>>>(WV, WVh, nw4);
    cvt_f32_f16<<<(nw4 + 255) / 256, 256, 0, stream>>>(WO, WOh, nw4);

    // ---- projections ----
    dim3 gg(Mrows / 128, Dc / 128);            // (32, 8)
    gemm_h<true><<<gg, 256, 0, stream>>>(Xh, WQh, (void*)Qh);
    gemm_h<true><<<gg, 256, 0, stream>>>(Xh, WKh, (void*)Kh);
    gemm_h<true><<<gg, 256, 0, stream>>>(Xh, WVh, (void*)Vh);

    // ---- RoPE on Q and K ----
    int npairs = (int)(szX / 2);
    rope_f16<<<(npairs + 255) / 256, 256, 0, stream>>>(Qh, pos, npairs);
    rope_f16<<<(npairs + 255) / 256, 256, 0, stream>>>(Kh, pos, npairs);

    // ---- causal flash attention ----
    flash_attn<<<dim3(Sc / 64, Hc, Bc), 128, 0, stream>>>(Qh, Kh, Vh, Oh);

    // ---- output projection (f32 result) ----
    gemm_h<false><<<gg, 256, 0, stream>>>(Oh, WOh, d_out);
}
